// Attention_27255862461085
// MI455X (gfx1250) — matrix operand order repaired and checked by reading, not compiled, not run on hardware
//
#include <hip/hip_runtime.h>

typedef __attribute__((ext_vector_type(16))) __bf16 v16bf;
typedef __attribute__((ext_vector_type(8)))  __bf16 v8bf;
typedef __attribute__((ext_vector_type(4)))  __bf16 v4bf;
typedef __attribute__((ext_vector_type(2)))  __bf16 v2bf;
typedef __attribute__((ext_vector_type(8)))  float  v8f;

#define WMMA_BF16(a, b, c) \
  __builtin_amdgcn_wmma_f32_16x16x32_bf16(false, (a), false, (b), (short)0, (c), false, false)

constexpr int B_  = 16;
constexpr int N_  = 2048;
constexpr int M_  = 2048;
constexpr int D_  = 128;
constexpr int DV_ = 128;
constexpr int KV  = 64;                // keys per step (4 score tiles)
constexpr int WAVES = 8;
constexpr int QW = 16;                 // query rows per wave
constexpr int QB = WAVES * QW;         // 128 query rows per block
constexpr float SCALE = 0.08838834764831845f;   // 1/sqrt(128)
constexpr float LOG2E = 1.4426950408889634f;
constexpr float CEXP  = SCALE * LOG2E;          // fold scale into exp2 domain

// padded LDS row lengths (bf16 elements) -> conflict-free strides
constexpr int KROW = D_ + 8;           // 136 (272 B rows)
constexpr int VROW = KV + 8;           // 72  (144 B rows)
constexpr int PROW = KV + 8;           // 72
constexpr int NSEG = (KV * D_) / (4 * 256);   // 8 float4 per thread per tensor

__device__ __forceinline__ v16bf ld2x8(const __bf16* p0, const __bf16* p1) {
  v8bf lo = *(const v8bf*)p0;
  v8bf hi = *(const v8bf*)p1;
  return __builtin_shufflevector(lo, hi, 0,1,2,3,4,5,6,7,8,9,10,11,12,13,14,15);
}

__device__ __forceinline__ v2bf cvt2(float x, float y) {
#if __has_builtin(__builtin_amdgcn_cvt_pk_bf16_f32)
  return __builtin_amdgcn_cvt_pk_bf16_f32(x, y);
#else
  v2bf r; r[0] = (__bf16)x; r[1] = (__bf16)y; return r;
#endif
}

__device__ __forceinline__ v4bf cvt4(float4 x) {
  v2bf lo = cvt2(x.x, x.y);
  v2bf hi = cvt2(x.z, x.w);
  return __builtin_shufflevector(lo, hi, 0,1,2,3);
}

__device__ __forceinline__ float fast_rcp(float x) {
#if __has_builtin(__builtin_amdgcn_rcpf)
  return __builtin_amdgcn_rcpf(x);
#else
  return 1.0f / x;
#endif
}

__global__ __launch_bounds__(256, 1)
void fa_fwd_bf16_wmma(const float* __restrict__ Q, const float* __restrict__ K,
                      const float* __restrict__ V, float* __restrict__ O)
{
  __shared__ __align__(32) __bf16 sK [KV * KROW];        // [key][feat]   ~17 KB
  __shared__ __align__(32) __bf16 sVt[DV_ * VROW];       // [dvfeat][key] ~18 KB
  __shared__ __align__(32) __bf16 sP [WAVES][QW * PROW]; // per-wave P    ~18 KB

  const int tid  = threadIdx.x;
  const int wave = tid >> 5;
  const int lane = tid & 31;
  const int half = lane >> 4;
  const int lc   = lane & 15;

  const int batch = blockIdx.x / (N_ / QB);
  const int qblk  = blockIdx.x % (N_ / QB);
  const int qrow0 = qblk * QB + wave * QW;

  const float* Qb = Q + (size_t)batch * N_ * D_;
  const float* Kb = K + (size_t)batch * M_ * D_;
  const float* Vb = V + (size_t)batch * M_ * DV_;
  float*       Ob = O + (size_t)batch * N_ * DV_;

  // ---- preload Q as 4 bf16 A-fragments (16x32, d=128) ----
  // A-layout: lane holds row lc; element e -> K = e + 8*((e>>3)+half)
  v16bf qa[4];
  {
    const float* qrow = Qb + (size_t)(qrow0 + lc) * D_;
    #pragma unroll
    for (int t = 0; t < 4; ++t) {
      const float* c0 = qrow + 32*t + 8*half;        // chunk1: K = 8h..8h+7
      const float* c1 = qrow + 32*t + 16 + 8*half;   // chunk2: K = 16+8h..+7
      v4bf l0 = cvt4(*(const float4*)(c0));
      v4bf l1 = cvt4(*(const float4*)(c0 + 4));
      v4bf h0 = cvt4(*(const float4*)(c1));
      v4bf h1 = cvt4(*(const float4*)(c1 + 4));
      v8bf lo = __builtin_shufflevector(l0, l1, 0,1,2,3,4,5,6,7);
      v8bf hi = __builtin_shufflevector(h0, h1, 0,1,2,3,4,5,6,7);
      qa[t] = __builtin_shufflevector(lo, hi, 0,1,2,3,4,5,6,7,8,9,10,11,12,13,14,15);
    }
  }

  v8f o_acc[8];
  #pragma unroll
  for (int nb = 0; nb < 8; ++nb)
    #pragma unroll
    for (int r = 0; r < 8; ++r) o_acc[nb][r] = 0.f;

  float m_run[8], l_run[8];
  #pragma unroll
  for (int r = 0; r < 8; ++r) { m_run[r] = -3.0e38f; l_run[r] = 0.f; }

  __bf16* sPw = &sP[wave][0];

  // per-thread slice of the 64x128 K/V tile: 8 float4 each
  float4 kreg[NSEG], vreg[NSEG];

  // prologue: fetch tile 0 and commit it to LDS
  #pragma unroll
  for (int i = 0; i < NSEG; ++i) {
    const int idx = (i * 256 + tid) * 4;
    kreg[i] = *(const float4*)(Kb + idx);
    vreg[i] = *(const float4*)(Vb + idx);
  }
  #pragma unroll
  for (int i = 0; i < NSEG; ++i) {
    const int idx = (i * 256 + tid) * 4;
    const int r = idx / D_, f = idx % D_;
    *(v4bf*)(sK + r*KROW + f) = cvt4(kreg[i]);
    v4bf vv = cvt4(vreg[i]);
    sVt[(f+0)*VROW + r] = vv[0];
    sVt[(f+1)*VROW + r] = vv[1];
    sVt[(f+2)*VROW + r] = vv[2];
    sVt[(f+3)*VROW + r] = vv[3];
  }

  #pragma unroll 1
  for (int j0 = 0; j0 < M_; j0 += KV) {
    const bool more = (j0 + KV) < M_;
    // ---- prefetch NEXT tile into registers (latency overlapped w/ compute) ----
    if (more) {
      const float* gk = Kb + (size_t)(j0 + KV) * D_;
      const float* gv = Vb + (size_t)(j0 + KV) * DV_;
      #pragma unroll
      for (int i = 0; i < NSEG; ++i) {
        const int idx = (i * 256 + tid) * 4;
        kreg[i] = *(const float4*)(gk + idx);
        vreg[i] = *(const float4*)(gv + idx);
      }
    }
    __syncthreads();   // LDS tile (stored last iteration) now visible

    // ---- S = Q * K^T : four 16x16 f32 tiles over 64 keys ----
    v8f s[4];
    #pragma unroll
    for (int n = 0; n < 4; ++n)
      #pragma unroll
      for (int r = 0; r < 8; ++r) s[n][r] = 0.f;
    #pragma unroll
    for (int t = 0; t < 4; ++t) {
      #pragma unroll
      for (int n = 0; n < 4; ++n) {
        const __bf16* kb = sK + (n*16 + lc)*KROW + 32*t + 8*half;
        s[n] = WMMA_BF16(qa[t], ld2x8(kb, kb + 16), s[n]);
      }
    }

    // ---- online softmax (row = r + 8*half, col = 16n + lc) ----
    float alpha[8];
    #pragma unroll
    for (int r = 0; r < 8; ++r) {
      float rm = fmaxf(fmaxf(s[0][r], s[1][r]), fmaxf(s[2][r], s[3][r]));
      rm = fmaxf(rm, __shfl_xor(rm, 1));
      rm = fmaxf(rm, __shfl_xor(rm, 2));
      rm = fmaxf(rm, __shfl_xor(rm, 4));
      rm = fmaxf(rm, __shfl_xor(rm, 8));     // masks <16 keep halves separate
      const float mnew = fmaxf(m_run[r], rm);
      alpha[r] = __builtin_amdgcn_exp2f((m_run[r] - mnew) * CEXP);
      m_run[r] = mnew;
      const float mc = mnew * CEXP;
      float rs = 0.f;
      #pragma unroll
      for (int n = 0; n < 4; ++n) {
        const float e = __builtin_amdgcn_exp2f(fmaf(s[n][r], CEXP, -mc));
        s[n][r] = e;
        rs += e;
      }
      rs += __shfl_xor(rs, 1);
      rs += __shfl_xor(rs, 2);
      rs += __shfl_xor(rs, 4);
      rs += __shfl_xor(rs, 8);
      l_run[r] = fmaf(alpha[r], l_run[r], rs);
    }
    #pragma unroll
    for (int nb = 0; nb < 8; ++nb)
      #pragma unroll
      for (int r = 0; r < 8; ++r) o_acc[nb][r] *= alpha[r];

    // ---- stage P -> LDS -> two A-fragments (keys 0-31, 32-63) ----
    #pragma unroll
    for (int r = 0; r < 8; ++r)
      #pragma unroll
      for (int n = 0; n < 4; ++n)
        sPw[(r + 8*half)*PROW + 16*n + lc] = (__bf16)s[n][r];
    asm volatile("s_wait_dscnt 0" ::: "memory");  // same-wave DS in-order
    const v16bf pa0 = ld2x8(sPw + lc*PROW + 8*half,
                            sPw + lc*PROW + 16 + 8*half);
    const v16bf pa1 = ld2x8(sPw + lc*PROW + 32 + 8*half,
                            sPw + lc*PROW + 48 + 8*half);

    // ---- O += P * V : 8 dv-blocks x 2 key-halves = 16 WMMAs ----
    #pragma unroll
    for (int nb = 0; nb < 8; ++nb) {
      const __bf16* vb = sVt + (nb*16 + lc)*VROW + 8*half;
      o_acc[nb] = WMMA_BF16(pa0, ld2x8(vb,      vb + 16), o_acc[nb]);
      o_acc[nb] = WMMA_BF16(pa1, ld2x8(vb + 32, vb + 48), o_acc[nb]);
    }

    __syncthreads();   // all waves done reading sK/sVt
    // ---- commit prefetched tile to LDS ----
    if (more) {
      #pragma unroll
      for (int i = 0; i < NSEG; ++i) {
        const int idx = (i * 256 + tid) * 4;
        const int r = idx / D_, f = idx % D_;
        *(v4bf*)(sK + r*KROW + f) = cvt4(kreg[i]);
        v4bf vv = cvt4(vreg[i]);
        sVt[(f+0)*VROW + r] = vv[0];
        sVt[(f+1)*VROW + r] = vv[1];
        sVt[(f+2)*VROW + r] = vv[2];
        sVt[(f+3)*VROW + r] = vv[3];
      }
    }
  }

  // ---- epilogue ----
  float inv_l[8];
  #pragma unroll
  for (int r = 0; r < 8; ++r) inv_l[r] = fast_rcp(l_run[r]);
  #pragma unroll
  for (int nb = 0; nb < 8; ++nb)
    #pragma unroll
    for (int r = 0; r < 8; ++r)
      Ob[(size_t)(qrow0 + r + 8*half) * DV_ + nb*16 + lc] = o_acc[nb][r] * inv_l[r];
}

extern "C" void kernel_launch(void* const* d_in, const int* in_sizes, int n_in,
                              void* d_out, int out_size, void* d_ws, size_t ws_size,
                              hipStream_t stream) {
  const float* q = (const float*)d_in[0];
  const float* k = (const float*)d_in[1];
  const float* v = (const float*)d_in[2];
  float* out = (float*)d_out;
  (void)in_sizes; (void)n_in; (void)out_size; (void)d_ws; (void)ws_size;
  dim3 grid(B_ * (N_ / QB));   // 16 batches * 16 query blocks = 256
  dim3 block(256);             // 8 wave32 per block
  fa_fwd_bf16_wmma<<<grid, block, 0, stream>>>(q, k, v, out);
}
